// SparseMoELayer_41798621724831
// MI455X (gfx1250) — matrix operand order repaired and checked by reading, not compiled, not run on hardware
//
#include <hip/hip_runtime.h>
#include <stdint.h>

#define TOK   8192
#define HID   2048
#define INTER 4096
#define NEXP  8
#define TOPK  2
#define AUX_COEF 0.01f
#define Z_COEF   0.001f

typedef __attribute__((ext_vector_type(16))) __bf16    bf16x16;
typedef __attribute__((ext_vector_type(8)))  float     f32x8;
typedef __attribute__((ext_vector_type(4)))  unsigned  u32x4;
typedef __attribute__((ext_vector_type(8)))  int       i32x8;
typedef __attribute__((ext_vector_type(4)))  int       i32x4;

union Frag { bf16x16 v; uint4 q[2]; };

__device__ __forceinline__ f32x8 wmma_bf16(const Frag& a, const Frag& b, f32x8 c) {
  return __builtin_amdgcn_wmma_f32_16x16x32_bf16(false, a.v, false, b.v, (short)0, c,
                                                 false, false);
}

__device__ __forceinline__ unsigned pack2(float a, float b) {
  union { __bf16 h[2]; unsigned u; } t;
  t.h[0] = (__bf16)a; t.h[1] = (__bf16)b;
  return t.u;
}

// ---- Tensor Data Mover: 2D bf16 tile (tile_rows x 32 elems) global -> LDS ----
// LDS destination uses the padded layout: 4 DWORDs of pad every 16 DWORDs
// (row stride 40 halfs), matching the WMMA fragment readers.
__device__ __forceinline__ void tdm_load_tile_bf16(unsigned lds_addr, const void* gaddr,
                                                   unsigned dim0_elems,   // tensor row len
                                                   unsigned dim1_rows,    // tensor #rows
                                                   unsigned stride_elems, // row stride
                                                   unsigned tile_rows) {
  unsigned long long ga = (unsigned long long)(uintptr_t)gaddr;
  u32x4 g0;
  g0[0] = 1u;                                   // count=1 (valid descriptor)
  g0[1] = lds_addr;                             // LDS byte address
  g0[2] = (unsigned)ga;                         // global addr [31:0]
  g0[3] = (unsigned)(ga >> 32) | (2u << 30);    // global addr [56:32] | type=2
  i32x8 g1;
  // data_size=2B(code1), pad_enable, pad_interval=16DW(code3), pad_amount=4DW(code3)
  g1[0] = (int)((1u << 16) | (1u << 20) | (3u << 22) | (3u << 25));
  g1[1] = (int)((dim0_elems & 0xFFFFu) << 16);                       // tensor_dim0 lo
  g1[2] = (int)((dim0_elems >> 16) | ((dim1_rows & 0xFFFFu) << 16)); // dim0 hi | dim1 lo
  g1[3] = (int)((dim1_rows >> 16) | (32u << 16));                    // dim1 hi | tile_dim0=32
  g1[4] = (int)(tile_rows & 0xFFFFu);                                // tile_dim1, tile_dim2=0
  g1[5] = (int)stride_elems;                                         // tensor_dim0_stride lo
  g1[6] = 0;
  g1[7] = 0;
  i32x4 z4 = {0, 0, 0, 0};
  i32x8 z8 = {0, 0, 0, 0, 0, 0, 0, 0};
  __builtin_amdgcn_tensor_load_to_lds(g0, g1, z4, z4, z8, 0);
}

__device__ __forceinline__ unsigned lds_off(const void* p) {
  // flat LDS address: low 32 bits are the LDS byte offset (aperture in high bits)
  return (unsigned)(uintptr_t)p;
}

// ---------------- zero output + counters ----------------
__global__ void k_zero(float* __restrict__ out, int n, int* __restrict__ cnt) {
  int stride = gridDim.x * blockDim.x;
  for (int i = blockIdx.x * blockDim.x + threadIdx.x; i < n; i += stride) out[i] = 0.f;
  if (blockIdx.x == 0 && threadIdx.x < NEXP) cnt[threadIdx.x] = 0;
}

// ---------------- fp32 -> bf16 copy (activations and weights) ----------------
__global__ void k_cvt(const float* __restrict__ x, __bf16* __restrict__ xb) {
  int i = blockIdx.x * blockDim.x + threadIdx.x;  // 4 elements per thread
  float4 v = ((const float4*)x)[i];
  uint2 p; p.x = pack2(v.x, v.y); p.y = pack2(v.z, v.w);
  ((uint2*)xb)[i] = p;
}

// ---------------- router: logits, softmax stats, top-2, gather lists ----------------
__global__ __launch_bounds__(256) void k_router(const float* __restrict__ x,
                                                const float* __restrict__ rw,
                                                float* __restrict__ probs,
                                                float* __restrict__ lse,
                                                int* __restrict__ cnt,
                                                int* __restrict__ list,
                                                float* __restrict__ wlist) {
  int t    = blockIdx.x * 8 + (threadIdx.x >> 5);   // one wave32 per token
  int lane = threadIdx.x & 31;
  const float* xr = x + (size_t)t * HID;

  float acc[NEXP];
#pragma unroll
  for (int e = 0; e < NEXP; e++) acc[e] = 0.f;

  for (int h = lane * 4; h < HID; h += 128) {
    float4 xv = *(const float4*)(xr + h);
#pragma unroll
    for (int e = 0; e < NEXP; e++) {
      float4 rv = *(const float4*)(rw + e * HID + h);
      acc[e] += xv.x * rv.x + xv.y * rv.y + xv.z * rv.z + xv.w * rv.w;
    }
  }
#pragma unroll
  for (int e = 0; e < NEXP; e++) {
#pragma unroll
    for (int m = 16; m >= 1; m >>= 1) acc[e] += __shfl_xor(acc[e], m, 32);
  }

  if (lane == 0) {
    float mx = acc[0];
#pragma unroll
    for (int e = 1; e < NEXP; e++) mx = fmaxf(mx, acc[e]);
    float p[NEXP], s = 0.f;
#pragma unroll
    for (int e = 0; e < NEXP; e++) { p[e] = __expf(acc[e] - mx); s += p[e]; }
    float inv = 1.f / s;
#pragma unroll
    for (int e = 0; e < NEXP; e++) probs[(size_t)t * NEXP + e] = p[e] * inv;
    lse[t] = mx + __logf(s);

    int e0 = 0;
#pragma unroll
    for (int e = 1; e < NEXP; e++) if (acc[e] > acc[e0]) e0 = e;
    int e1 = (e0 == 0) ? 1 : 0;
#pragma unroll
    for (int e = 0; e < NEXP; e++) if (e != e0 && acc[e] > acc[e1]) e1 = e;

    float d  = __expf(acc[e1] - acc[e0]);       // l1 <= l0
    float w0 = 1.f / (1.f + d);
    float w1 = d / (1.f + d);

    int p0 = atomicAdd(&cnt[e0], 1);
    list[e0 * TOK + p0]  = t;  wlist[e0 * TOK + p0] = w0;
    int p1 = atomicAdd(&cnt[e1], 1);
    list[e1 * TOK + p1]  = t;  wlist[e1 * TOK + p1] = w1;
  }
}

// ---------------- deterministic reductions for aux losses ----------------
__global__ __launch_bounds__(256) void k_reduce(const float* __restrict__ probs,
                                                const float* __restrict__ lse,
                                                float* __restrict__ psum,
                                                float* __restrict__ lsum) {
  __shared__ float red[256];
  int b = blockIdx.x, tid = threadIdx.x;
  float s = 0.f;
  int t0 = tid * (TOK / 256);
  if (b < NEXP) {
    for (int i = 0; i < TOK / 256; i++) s += probs[(size_t)(t0 + i) * NEXP + b];
  } else {
    for (int i = 0; i < TOK / 256; i++) s += lse[t0 + i];
  }
  red[tid] = s; __syncthreads();
  for (int st = 128; st > 0; st >>= 1) {
    if (tid < st) red[tid] += red[tid + st];
    __syncthreads();
  }
  if (tid == 0) { if (b < NEXP) psum[b] = red[0]; else lsum[0] = red[0]; }
}

// ---------------- exclusive scan of counts + final loss scalar ----------------
__global__ void k_finalize(const int* __restrict__ cnt, const float* __restrict__ psum,
                           const float* __restrict__ lsum, int* __restrict__ base,
                           float* __restrict__ out_loss) {
  if (blockIdx.x == 0 && threadIdx.x == 0) {
    int a = 0; float aux = 0.f;
    for (int e = 0; e < NEXP; e++) {
      base[e] = a; a += cnt[e];
      aux += ((float)cnt[e] / (float)(TOK * TOPK)) * (psum[e] / (float)TOK);
    }
    aux *= (float)NEXP * AUX_COEF;
    out_loss[0] = aux + (lsum[0] / (float)TOK) * Z_COEF;
  }
}

// ---------------- fused gate/up GEMM + SiLU epilogue (WMMA + TDM) ----------------
__global__ __launch_bounds__(128) void k_gateup(const __bf16* __restrict__ xb,
                                                const __bf16* __restrict__ gwb,
                                                const __bf16* __restrict__ uwb,
                                                const int* __restrict__ cnt,
                                                const int* __restrict__ base,
                                                const int* __restrict__ list,
                                                __bf16* __restrict__ hbuf) {
  const int e  = blockIdx.z;
  const int m0 = blockIdx.y * 64;
  const int n0 = blockIdx.x * 64;
  const int c  = cnt[e];
  if (m0 >= c) return;
  const int b0 = base[e];

  __shared__ __attribute__((aligned(16))) __bf16 sA[2][64 * 40];
  __shared__ __attribute__((aligned(16))) __bf16 sG[2][64 * 40];
  __shared__ __attribute__((aligned(16))) __bf16 sU[2][64 * 40];
  __shared__ int sTok[64];

  const int tid = threadIdx.x;
  const int lane = tid & 31, wave = tid >> 5;
  const int wm = wave >> 1, wn = wave & 1;
  const int l2 = lane & 15, hi = lane >> 4;

  if (tid < 64) {
    int pos = m0 + tid;
    sTok[tid] = list[(size_t)e * TOK + ((pos < c) ? pos : 0)];
  }
  __syncthreads();

  const __bf16* gb = gwb + ((size_t)e * INTER + n0) * HID;
  const __bf16* ub = uwb + ((size_t)e * INTER + n0) * HID;

  f32x8 cg[2][2] = {};
  f32x8 cu[2][2] = {};

  const int kt = HID / 32;
  uint4 areg[2];

  // ---- prologue: stage k=0 tiles ----
  if (wave == 0) {
    tdm_load_tile_bf16(lds_off(&sG[0][0]), gb, HID, INTER, HID, 64);
    tdm_load_tile_bf16(lds_off(&sU[0][0]), ub, HID, INTER, HID, 64);
  }
#pragma unroll
  for (int it = 0; it < 2; it++) {
    int ch = tid + it * 128;
    int row = ch >> 2, ko = (ch & 3) * 8;
    areg[it] = *(const uint4*)(xb + (size_t)sTok[row] * HID + ko);
  }
#pragma unroll
  for (int it = 0; it < 2; it++) {
    int ch = tid + it * 128;
    int row = ch >> 2, ko = (ch & 3) * 8;
    *(uint4*)&sA[0][row * 40 + ko] = areg[it];
  }
  if (wave == 0) __builtin_amdgcn_s_wait_tensorcnt(0);
  __syncthreads();

  for (int k = 0; k < kt; k++) {
    const int cur = k & 1, nxt = cur ^ 1;
    const bool more = (k + 1) < kt;
    const int k1 = (k + 1) * 32;

    if (more) {
      if (wave == 0) {
        tdm_load_tile_bf16(lds_off(&sG[nxt][0]), gb + k1, HID, INTER, HID, 64);
        tdm_load_tile_bf16(lds_off(&sU[nxt][0]), ub + k1, HID, INTER, HID, 64);
      }
#pragma unroll
      for (int it = 0; it < 2; it++) {
        int ch = tid + it * 128;
        int row = ch >> 2, ko = (ch & 3) * 8;
        areg[it] = *(const uint4*)(xb + (size_t)sTok[row] * HID + (k1 + ko));
      }
    }

    Frag fa[2], fg[2], fu[2];
    const int kb = hi * 8;    // A: K 0-7/16-23 (lo lanes), 8-15/24-31 (hi lanes)
    const int kc = hi * 8;    // B: K 0-7/16-23 (lo lanes), 8-15/24-31 (hi lanes), col = l2
#pragma unroll
    for (int mi = 0; mi < 2; mi++) {
      int r = wm * 32 + mi * 16 + l2;
      fa[mi].q[0] = *(const uint4*)&sA[cur][r * 40 + kb];
      fa[mi].q[1] = *(const uint4*)&sA[cur][r * 40 + kb + 16];
    }
#pragma unroll
    for (int ni = 0; ni < 2; ni++) {
      int r = wn * 32 + ni * 16 + l2;
      fg[ni].q[0] = *(const uint4*)&sG[cur][r * 40 + kc];
      fg[ni].q[1] = *(const uint4*)&sG[cur][r * 40 + kc + 16];
      fu[ni].q[0] = *(const uint4*)&sU[cur][r * 40 + kc];
      fu[ni].q[1] = *(const uint4*)&sU[cur][r * 40 + kc + 16];
    }
#pragma unroll
    for (int mi = 0; mi < 2; mi++) {
#pragma unroll
      for (int ni = 0; ni < 2; ni++) {
        cg[mi][ni] = wmma_bf16(fa[mi], fg[ni], cg[mi][ni]);
        cu[mi][ni] = wmma_bf16(fa[mi], fu[ni], cu[mi][ni]);
      }
    }

    if (more) {
#pragma unroll
      for (int it = 0; it < 2; it++) {
        int ch = tid + it * 128;
        int row = ch >> 2, ko = (ch & 3) * 8;
        *(uint4*)&sA[nxt][row * 40 + ko] = areg[it];
      }
    }
    if (wave == 0) __builtin_amdgcn_s_wait_tensorcnt(0);
    __syncthreads();
  }

  // SiLU(gate) * up -> bf16 hbuf rows (compact per-expert)
#pragma unroll
  for (int mi = 0; mi < 2; mi++) {
#pragma unroll
    for (int ni = 0; ni < 2; ni++) {
#pragma unroll
      for (int r = 0; r < 8; r++) {
        int ml = wm * 32 + mi * 16 + hi * 8 + r;
        int pos = m0 + ml;
        if (pos < c) {
          float g = cg[mi][ni][r];
          float u = cu[mi][ni][r];
          float hval = g * u * __builtin_amdgcn_rcpf(1.f + __expf(-g));
          hbuf[(size_t)(b0 + pos) * INTER + (n0 + wn * 32 + ni * 16 + l2)] = (__bf16)hval;
        }
      }
    }
  }
}

// ---------------- down GEMM (WMMA + TDM for both tiles) + atomic combine ----------------
__global__ __launch_bounds__(128) void k_down(const __bf16* __restrict__ hbuf,
                                              const __bf16* __restrict__ dwb,
                                              const int* __restrict__ cnt,
                                              const int* __restrict__ base,
                                              const int* __restrict__ list,
                                              const float* __restrict__ wlist,
                                              float* __restrict__ out) {
  const int e  = blockIdx.z;
  const int m0 = blockIdx.y * 64;
  const int n0 = blockIdx.x * 64;     // over HID
  const int c  = cnt[e];
  if (m0 >= c) return;
  const int b0 = base[e];

  __shared__ __attribute__((aligned(16))) __bf16 sA[2][64 * 40];
  __shared__ __attribute__((aligned(16))) __bf16 sB[2][64 * 40];
  __shared__ int   sTok[64];
  __shared__ float sW[64];

  const int tid = threadIdx.x;
  const int lane = tid & 31, wave = tid >> 5;
  const int wm = wave >> 1, wn = wave & 1;
  const int l2 = lane & 15, hi = lane >> 4;

  if (tid < 64) {
    int pos = m0 + tid;
    bool ok = pos < c;
    sTok[tid] = ok ? list[(size_t)e * TOK + pos] : 0;
    sW[tid]   = ok ? wlist[(size_t)e * TOK + pos] : 0.f;
  }

  const __bf16* ab = hbuf + (size_t)(b0 + m0) * INTER;   // 64 sequential rows (padded alloc)
  const __bf16* db = dwb + ((size_t)e * HID + n0) * INTER;

  f32x8 cc[2][2] = {};
  const int kt = INTER / 32;

  if (wave == 0) {
    tdm_load_tile_bf16(lds_off(&sA[0][0]), ab, INTER, TOK * TOPK + 64, INTER, 64);
    tdm_load_tile_bf16(lds_off(&sB[0][0]), db, INTER, HID, INTER, 64);
    __builtin_amdgcn_s_wait_tensorcnt(0);
  }
  __syncthreads();

  for (int k = 0; k < kt; k++) {
    const int cur = k & 1, nxt = cur ^ 1;
    const bool more = (k + 1) < kt;
    const int k1 = (k + 1) * 32;

    if (more && wave == 0) {
      tdm_load_tile_bf16(lds_off(&sA[nxt][0]), ab + k1, INTER, TOK * TOPK + 64, INTER, 64);
      tdm_load_tile_bf16(lds_off(&sB[nxt][0]), db + k1, INTER, HID, INTER, 64);
    }

    Frag fa[2], fb[2];
    const int kb = hi * 8;
    const int kc = hi * 8;
#pragma unroll
    for (int mi = 0; mi < 2; mi++) {
      int r = wm * 32 + mi * 16 + l2;
      fa[mi].q[0] = *(const uint4*)&sA[cur][r * 40 + kb];
      fa[mi].q[1] = *(const uint4*)&sA[cur][r * 40 + kb + 16];
    }
#pragma unroll
    for (int ni = 0; ni < 2; ni++) {
      int r = wn * 32 + ni * 16 + l2;
      fb[ni].q[0] = *(const uint4*)&sB[cur][r * 40 + kc];
      fb[ni].q[1] = *(const uint4*)&sB[cur][r * 40 + kc + 16];
    }
#pragma unroll
    for (int mi = 0; mi < 2; mi++) {
#pragma unroll
      for (int ni = 0; ni < 2; ni++) {
        cc[mi][ni] = wmma_bf16(fa[mi], fb[ni], cc[mi][ni]);
      }
    }

    if (wave == 0) __builtin_amdgcn_s_wait_tensorcnt(0);
    __syncthreads();
  }

  // out[token] += w * down  (exactly 2 commutative f32 atomic adds per element)
#pragma unroll
  for (int mi = 0; mi < 2; mi++) {
#pragma unroll
    for (int ni = 0; ni < 2; ni++) {
#pragma unroll
      for (int r = 0; r < 8; r++) {
        int ml = wm * 32 + mi * 16 + hi * 8 + r;
        int pos = m0 + ml;
        if (pos < c) {
          int tok = sTok[ml];
          float w = sW[ml];
          atomicAdd(out + (size_t)tok * HID + (n0 + wn * 32 + ni * 16 + l2),
                    cc[mi][ni][r] * w);
        }
      }
    }
  }
}

extern "C" void kernel_launch(void* const* d_in, const int* in_sizes, int n_in,
                              void* d_out, int out_size, void* d_ws, size_t ws_size,
                              hipStream_t stream) {
  (void)in_sizes; (void)n_in; (void)out_size; (void)ws_size;
  const float* x  = (const float*)d_in[0];
  const float* rw = (const float*)d_in[1];
  const float* gw = (const float*)d_in[2];
  const float* uw = (const float*)d_in[3];
  const float* dw = (const float*)d_in[4];
  float* out = (float*)d_out;

  char* ws = (char*)d_ws;
  size_t off = 0;
  auto alloc = [&](size_t bytes) -> char* {
    char* p = ws + off;
    off += (bytes + 255) & ~(size_t)255;
    return p;
  };
  const size_t WELEM = (size_t)NEXP * INTER * HID;                   // 67,108,864
  __bf16* xb    = (__bf16*)alloc((size_t)TOK * HID * 2);             // 32 MB
  __bf16* gwb   = (__bf16*)alloc(WELEM * 2);                         // 134 MB
  __bf16* uwb   = (__bf16*)alloc(WELEM * 2);                         // 134 MB
  __bf16* dwb   = (__bf16*)alloc(WELEM * 2);                         // 134 MB
  __bf16* hbuf  = (__bf16*)alloc((size_t)(TOK * TOPK + 64) * INTER * 2); // 128.5 MB
  float*  probs = (float*) alloc((size_t)TOK * NEXP * 4);
  float*  lse   = (float*) alloc((size_t)TOK * 4);
  int*    cnt   = (int*)   alloc(NEXP * 4);
  int*    basep = (int*)   alloc(NEXP * 4);
  int*    list  = (int*)   alloc((size_t)NEXP * TOK * 4);
  float*  wlist = (float*) alloc((size_t)NEXP * TOK * 4);
  float*  psum  = (float*) alloc(NEXP * 4);
  float*  lsum  = (float*) alloc(4);

  k_zero<<<2048, 256, 0, stream>>>(out, TOK * HID + 1, cnt);
  k_cvt<<<(TOK * HID / 4) / 256, 256, 0, stream>>>(x, xb);
  k_cvt<<<(int)(WELEM / 4 / 256), 256, 0, stream>>>(gw, gwb);
  k_cvt<<<(int)(WELEM / 4 / 256), 256, 0, stream>>>(uw, uwb);
  k_cvt<<<(int)(WELEM / 4 / 256), 256, 0, stream>>>(dw, dwb);
  k_router<<<TOK / 8, 256, 0, stream>>>(x, rw, probs, lse, cnt, list, wlist);
  k_reduce<<<NEXP + 1, 256, 0, stream>>>(probs, lse, psum, lsum);
  k_finalize<<<1, 32, 0, stream>>>(cnt, psum, lsum, basep, out + (size_t)TOK * HID);
  k_gateup<<<dim3(INTER / 64, TOK / 64, NEXP), 128, 0, stream>>>(xb, gwb, uwb, cnt, basep,
                                                                 list, hbuf);
  k_down<<<dim3(HID / 64, TOK / 64, NEXP), 128, 0, stream>>>(hbuf, dwb, cnt, basep, list,
                                                             wlist, out);
}
